// PatchEmbed_37134287241632
// MI455X (gfx1250) — compile-verified, matrix operand order checked by reading, not run on hardware
//
#include <hip/hip_runtime.h>
#include <hip/hip_fp16.h>

#define B_   8
#define N_   32768
#define G_   512
#define K_   64
#define CIN_ 6
#define H_   448
#define OUT_ 256

typedef __attribute__((ext_vector_type(16))) _Float16 v16h;
typedef __attribute__((ext_vector_type(8)))  float    v8f;

// ---------- monotone float <-> uint for atomicMax-based float max ----------
__device__ __forceinline__ unsigned f2u(float f) {
  unsigned u = __float_as_uint(f);
  return (u & 0x80000000u) ? ~u : (u | 0x80000000u);
}
__device__ __forceinline__ float u2f(unsigned u) {
  return __uint_as_float((u & 0x80000000u) ? (u ^ 0x80000000u) : ~u);
}

// ---------- WMMA A-fragment (16x32 f16) element offset ----------
// Tiles stored [kt][mt 0..3][lane 0..31][16 halves].  Per ISA 16-bit A layout:
//  lanes 0-15 hold K=0..7 (halves 0..7) and K=16..23 (halves 8..15)
//  lanes 16-31 hold K=8..15 and K=24..31
__device__ __forceinline__ int afrag_elem(int kt, int mt, int mr, int kr) {
  int lane = mr + ((kr & 8) ? 16 : 0);
  int h    = (kr & 7) + ((kr & 16) ? 8 : 0);
  return ((kt * 4 + mt) * 32 + lane) * 16 + h;
}

// =====================================================================
// 1) Weight pack: f32 [K,N] row-major -> f16 fragment order [nt][kt][lane][16]
//    B-fragment mirrors A layout with N in place of M.
// =====================================================================
__device__ __forceinline__ void pack_one(const float* __restrict__ W,
                                         _Float16* __restrict__ dst,
                                         int Kdim, int Ndim, int e) {
  int k = e / Ndim, n = e % Ndim;
  int kt = k >> 5, kr = k & 31, nt = n >> 4, nr = n & 15;
  int lane = nr + ((kr & 8) ? 16 : 0);
  int h    = (kr & 7) + ((kr & 16) ? 8 : 0);
  int KT = Kdim >> 5;
  dst[((nt * KT + kt) * 32 + lane) * 16 + h] = (_Float16)W[e];
}

__global__ __launch_bounds__(256) void pack_kernel(const float* __restrict__ W2,
                                                   const float* __restrict__ W3,
                                                   const float* __restrict__ W4,
                                                   _Float16* __restrict__ wpack) {
  int e = blockIdx.x * 256 + threadIdx.x;
  const int S2 = H_ * H_;         // 200704
  const int S3 = 2 * H_ * H_;     // 401408
  const int S4 = H_ * OUT_;       // 114688
  if (e < S2)                pack_one(W2, wpack,            H_,     H_,   e);
  else if (e < S2 + S3)      pack_one(W3, wpack + S2,       2 * H_, H_,   e - S2);
  else if (e < S2 + S3 + S4) pack_one(W4, wpack + S2 + S3,  H_,     OUT_, e - S2 - S3);
}

// =====================================================================
// 2) Farthest point sampling: one block per batch, sequential over G.
//    min-distance kept in registers (32 per thread), argmax via LDS tree.
// =====================================================================
__global__ __launch_bounds__(1024) void fps_kernel(const float* __restrict__ coords,
                                                   int* __restrict__ fps_idx,
                                                   float* __restrict__ centers) {
  __shared__ float s_val[1024];
  __shared__ int   s_idx[1024];
  __shared__ float s_c[3];
  const int b = blockIdx.x, tid = threadIdx.x;
  const float* cb = coords + (size_t)b * N_ * 3;
  float mind[32];
#pragma unroll
  for (int j = 0; j < 32; ++j) mind[j] = 1e10f;
  int last = 0;
  for (int g = 0; g < G_; ++g) {
    if (tid == 0) {
      fps_idx[b * G_ + g] = last;
      float x = cb[last * 3 + 0], y = cb[last * 3 + 1], z = cb[last * 3 + 2];
      centers[((size_t)b * G_ + g) * 3 + 0] = x;
      centers[((size_t)b * G_ + g) * 3 + 1] = y;
      centers[((size_t)b * G_ + g) * 3 + 2] = z;
      s_c[0] = x; s_c[1] = y; s_c[2] = z;
    }
    __syncthreads();
    const float cx = s_c[0], cy = s_c[1], cz = s_c[2];
    float bv = -1.0f; int bi = 0;
#pragma unroll 4
    for (int j = 0; j < 32; ++j) {
      int p = tid + j * 1024;
      float dx = cb[p * 3 + 0] - cx, dy = cb[p * 3 + 1] - cy, dz = cb[p * 3 + 2] - cz;
      float d = dx * dx + dy * dy + dz * dz;
      float m = fminf(mind[j], d);
      mind[j] = m;
      if (m > bv) { bv = m; bi = p; }
    }
    s_val[tid] = bv; s_idx[tid] = bi;
    __syncthreads();
    for (int s = 512; s >= 1; s >>= 1) {
      if (tid < s) {
        float ov = s_val[tid + s];
        if (ov > s_val[tid] || (ov == s_val[tid] && s_idx[tid + s] < s_idx[tid])) {
          s_val[tid] = ov; s_idx[tid] = s_idx[tid + s];
        }
      }
      __syncthreads();
    }
    last = s_idx[0];
    __syncthreads();
  }
}

// =====================================================================
// 3) KNN: one block per (b,g); full distance array staged in dynamic LDS
//    (128KB, legal under CDNA5's 320KB/WGP); 64 argmin passes -> sorted knn.
// =====================================================================
__global__ __launch_bounds__(256) void knn_kernel(const float* __restrict__ coords,
                                                  const int* __restrict__ fps_idx,
                                                  int* __restrict__ knn_i,
                                                  float* __restrict__ knn_f) {
  extern __shared__ float sm[];
  float* d2 = sm;                         // N_ floats
  float* rv = sm + N_;                    // 256
  int*   ri = (int*)(sm + N_ + 256);      // 256
  const int bg = blockIdx.x, tid = threadIdx.x;
  const int b = bg / G_;
  const float* cb = coords + (size_t)b * N_ * 3;
  const int ci = fps_idx[bg];
  const float cx = cb[ci * 3 + 0], cy = cb[ci * 3 + 1], cz = cb[ci * 3 + 2];
  for (int p = tid; p < N_; p += 256) {
    float dx = cb[p * 3 + 0] - cx, dy = cb[p * 3 + 1] - cy, dz = cb[p * 3 + 2] - cz;
    d2[p] = dx * dx + dy * dy + dz * dz;
  }
  __syncthreads();
  for (int kk = 0; kk < K_; ++kk) {
    float bv = 3.0e38f; int bi = N_;
    for (int p = tid; p < N_; p += 256) {
      float v = d2[p];
      if (v < bv) { bv = v; bi = p; }
    }
    rv[tid] = bv; ri[tid] = bi;
    __syncthreads();
    for (int s = 128; s >= 1; s >>= 1) {
      if (tid < s) {
        if (rv[tid + s] < rv[tid] || (rv[tid + s] == rv[tid] && ri[tid + s] < ri[tid])) {
          rv[tid] = rv[tid + s]; ri[tid] = ri[tid + s];
        }
      }
      __syncthreads();
    }
    int w = ri[0];
    if (tid == 0) {
      knn_i[(size_t)bg * K_ + kk] = w;
      knn_f[(size_t)bg * K_ + kk] = (float)w;
      d2[w] = 3.0e38f;
    }
    __syncthreads();
  }
}

// =====================================================================
// 4) Fused patch encoder: one block (8 wave32) per patch.
//    layer1 (6->448) on VALU into f16 A-fragments in LDS;
//    layers 2/3/4 as 2x2 register-tiled WMMA f32_16x16x32_f16
//    (4 WMMAs per 2 A + 2 B fragment loads); max-pools via LDS atomicMax.
//    Dynamic LDS = 176384 B.
// =====================================================================
__global__ __launch_bounds__(256)
void patch_mlp_kernel(const float* __restrict__ coords, const float* __restrict__ feats,
                      const float* __restrict__ W1, const float* __restrict__ b1,
                      const float* __restrict__ b2, const float* __restrict__ b3,
                      const float* __restrict__ b4,
                      const int* __restrict__ fps_idx, const int* __restrict__ knn,
                      const _Float16* __restrict__ wpack, float* __restrict__ emb) {
  extern __shared__ char smraw[];
  _Float16* h1f   = (_Float16*)smraw;             // 57344 B  (h1, reused as h3)
  _Float16* catf  = (_Float16*)(smraw + 57344);   // 114688 B (cat = [g | x2])
  unsigned* gmaxU = (unsigned*)(smraw + 172032);  // 448 * 4
  unsigned* embU  = (unsigned*)(smraw + 173824);  // 256 * 4
  float*    pf    = (float*)(smraw + 174848);     // 64 * 6 * 4

  const int bg = blockIdx.x, tid = threadIdx.x;
  const int b = bg / G_;
  const int wave = tid >> 5, lane = tid & 31;
  const float* cb = coords + (size_t)b * N_ * 3;
  const float* fb = feats  + (size_t)b * N_ * 3;

  for (int n = tid; n < H_;   n += 256) gmaxU[n] = f2u(-3.0e38f);
  for (int n = tid; n < OUT_; n += 256) embU[n]  = f2u(-3.0e38f);

  // ---- gather patch features (rel coords + feats) ----
  if (tid < K_) {
    const int ci = fps_idx[bg];
    const float cx = cb[ci * 3 + 0], cy = cb[ci * 3 + 1], cz = cb[ci * 3 + 2];
    const int p = knn[(size_t)bg * K_ + tid];
    pf[tid * 6 + 0] = cb[p * 3 + 0] - cx;
    pf[tid * 6 + 1] = cb[p * 3 + 1] - cy;
    pf[tid * 6 + 2] = cb[p * 3 + 2] - cz;
    pf[tid * 6 + 3] = fb[p * 3 + 0];
    pf[tid * 6 + 4] = fb[p * 3 + 1];
    pf[tid * 6 + 5] = fb[p * 3 + 2];
  }
  __syncthreads();

  // ---- layer1: relu(pf @ W1 + b1) -> h1 fragments (f16) ----
  for (int i = tid; i < K_ * H_; i += 256) {
    int m = i / H_, c = i % H_;
    float acc = b1[c];
#pragma unroll
    for (int j = 0; j < 6; ++j) acc += pf[m * 6 + j] * W1[j * H_ + c];
    acc = fmaxf(acc, 0.0f);
    h1f[afrag_elem(c >> 5, m >> 4, m & 15, c & 31)] = (_Float16)acc;
  }
  __syncthreads();

  // ================= layer2: x2 = h1 @ W2 + b2 =================
  // 2x2 macro-tiles: (2 m-tiles) x (14 n-tile-pairs) = 28 macro-tiles
  {
    const _Float16* Wp = wpack;
    const int KT = 14, NT2 = 14;
    for (int t = wave; t < 2 * NT2; t += 8) {
      int mtp = t & 1, ntp = t >> 1;
      v8f acc00 = {}, acc01 = {}, acc10 = {}, acc11 = {};
      const _Float16* a_base = h1f + ((size_t)mtp * 2 * 32 + lane) * 16;
      const _Float16* b_base = Wp + (((size_t)ntp * 2 * KT) * 32 + lane) * 16;
      for (int kt = 0; kt < KT; ++kt) {
        v16h a0 = *(const v16h*)(a_base + (size_t)kt * 4 * 512);
        v16h a1 = *(const v16h*)(a_base + (size_t)kt * 4 * 512 + 512);
        v16h b0 = *(const v16h*)(b_base + (size_t)kt * 512);
        v16h b1v = *(const v16h*)(b_base + ((size_t)KT + kt) * 512);
        if (kt + 1 < KT)
          __builtin_prefetch(b_base + (size_t)(kt + 1) * 512, 0, 2);
        acc00 = __builtin_amdgcn_wmma_f32_16x16x32_f16(false, a0, false, b0,  (short)0, acc00, false, false);
        acc01 = __builtin_amdgcn_wmma_f32_16x16x32_f16(false, a0, false, b1v, (short)0, acc01, false, false);
        acc10 = __builtin_amdgcn_wmma_f32_16x16x32_f16(false, a1, false, b0,  (short)0, acc10, false, false);
        acc11 = __builtin_amdgcn_wmma_f32_16x16x32_f16(false, a1, false, b1v, (short)0, acc11, false, false);
      }
      v8f accs[2][2] = {{acc00, acc01}, {acc10, acc11}};
#pragma unroll
      for (int i = 0; i < 2; ++i) {
#pragma unroll
        for (int j = 0; j < 2; ++j) {
          int nc = (ntp * 2 + j) * 16 + (lane & 15);
          float bb = b2[nc];
          int mbase = (mtp * 2 + i) * 16 + ((lane & 16) ? 8 : 0);
#pragma unroll
          for (int v = 0; v < 8; ++v) {
            float val = accs[i][j][v] + bb;
            atomicMax(&gmaxU[nc], f2u(val));
            int m = mbase + v;
            int c = 448 + nc;
            catf[afrag_elem(c >> 5, m >> 4, m & 15, c & 31)] = (_Float16)val;
          }
        }
      }
    }
  }
  __syncthreads();

  // ---- broadcast global max into cat[:, :448] ----
  for (int i = tid; i < K_ * H_; i += 256) {
    int m = i / H_, c = i % H_;
    float gv = u2f(gmaxU[c]);
    catf[afrag_elem(c >> 5, m >> 4, m & 15, c & 31)] = (_Float16)gv;
  }
  __syncthreads();

  // ================= layer3: h3 = relu(cat @ W3 + b3) -> h1f =================
  {
    const _Float16* Wp = wpack + H_ * H_;
    const int KT = 28, NT2 = 14;
    for (int t = wave; t < 2 * NT2; t += 8) {
      int mtp = t & 1, ntp = t >> 1;
      v8f acc00 = {}, acc01 = {}, acc10 = {}, acc11 = {};
      const _Float16* a_base = catf + ((size_t)mtp * 2 * 32 + lane) * 16;
      const _Float16* b_base = Wp + (((size_t)ntp * 2 * KT) * 32 + lane) * 16;
      for (int kt = 0; kt < KT; ++kt) {
        v16h a0 = *(const v16h*)(a_base + (size_t)kt * 4 * 512);
        v16h a1 = *(const v16h*)(a_base + (size_t)kt * 4 * 512 + 512);
        v16h b0 = *(const v16h*)(b_base + (size_t)kt * 512);
        v16h b1v = *(const v16h*)(b_base + ((size_t)KT + kt) * 512);
        if (kt + 1 < KT)
          __builtin_prefetch(b_base + (size_t)(kt + 1) * 512, 0, 2);
        acc00 = __builtin_amdgcn_wmma_f32_16x16x32_f16(false, a0, false, b0,  (short)0, acc00, false, false);
        acc01 = __builtin_amdgcn_wmma_f32_16x16x32_f16(false, a0, false, b1v, (short)0, acc01, false, false);
        acc10 = __builtin_amdgcn_wmma_f32_16x16x32_f16(false, a1, false, b0,  (short)0, acc10, false, false);
        acc11 = __builtin_amdgcn_wmma_f32_16x16x32_f16(false, a1, false, b1v, (short)0, acc11, false, false);
      }
      v8f accs[2][2] = {{acc00, acc01}, {acc10, acc11}};
#pragma unroll
      for (int i = 0; i < 2; ++i) {
#pragma unroll
        for (int j = 0; j < 2; ++j) {
          int nc = (ntp * 2 + j) * 16 + (lane & 15);
          float bb = b3[nc];
          int mbase = (mtp * 2 + i) * 16 + ((lane & 16) ? 8 : 0);
#pragma unroll
          for (int v = 0; v < 8; ++v) {
            float val = fmaxf(accs[i][j][v] + bb, 0.0f);
            int m = mbase + v;
            h1f[afrag_elem(nc >> 5, m >> 4, m & 15, nc & 31)] = (_Float16)val;
          }
        }
      }
    }
  }
  __syncthreads();

  // ================= layer4: x4 = h3 @ W4 + b4 ; emb = row max =================
  {
    const _Float16* Wp = wpack + H_ * H_ + 2 * H_ * H_;
    const int KT = 14, NT2 = 8;
    for (int t = wave; t < 2 * NT2; t += 8) {
      int mtp = t & 1, ntp = t >> 1;
      v8f acc00 = {}, acc01 = {}, acc10 = {}, acc11 = {};
      const _Float16* a_base = h1f + ((size_t)mtp * 2 * 32 + lane) * 16;
      const _Float16* b_base = Wp + (((size_t)ntp * 2 * KT) * 32 + lane) * 16;
      for (int kt = 0; kt < KT; ++kt) {
        v16h a0 = *(const v16h*)(a_base + (size_t)kt * 4 * 512);
        v16h a1 = *(const v16h*)(a_base + (size_t)kt * 4 * 512 + 512);
        v16h b0 = *(const v16h*)(b_base + (size_t)kt * 512);
        v16h b1v = *(const v16h*)(b_base + ((size_t)KT + kt) * 512);
        acc00 = __builtin_amdgcn_wmma_f32_16x16x32_f16(false, a0, false, b0,  (short)0, acc00, false, false);
        acc01 = __builtin_amdgcn_wmma_f32_16x16x32_f16(false, a0, false, b1v, (short)0, acc01, false, false);
        acc10 = __builtin_amdgcn_wmma_f32_16x16x32_f16(false, a1, false, b0,  (short)0, acc10, false, false);
        acc11 = __builtin_amdgcn_wmma_f32_16x16x32_f16(false, a1, false, b1v, (short)0, acc11, false, false);
      }
      v8f accs[2][2] = {{acc00, acc01}, {acc10, acc11}};
#pragma unroll
      for (int i = 0; i < 2; ++i) {
#pragma unroll
        for (int j = 0; j < 2; ++j) {
          int nc = (ntp * 2 + j) * 16 + (lane & 15);
          float bb = b4[nc];
#pragma unroll
          for (int v = 0; v < 8; ++v) {
            float val = accs[i][j][v] + bb;
            atomicMax(&embU[nc], f2u(val));
          }
        }
      }
    }
  }
  __syncthreads();

  for (int n = tid; n < OUT_; n += 256)
    emb[(size_t)bg * OUT_ + n] = u2f(embU[n]);
}

// =====================================================================
// launch
// =====================================================================
extern "C" void kernel_launch(void* const* d_in, const int* in_sizes, int n_in,
                              void* d_out, int out_size, void* d_ws, size_t ws_size,
                              hipStream_t stream) {
  (void)in_sizes; (void)n_in; (void)out_size; (void)ws_size;
  const float* coords = (const float*)d_in[0];
  const float* feats  = (const float*)d_in[1];
  const float* W1 = (const float*)d_in[2];
  const float* b1 = (const float*)d_in[3];
  const float* W2 = (const float*)d_in[4];
  const float* b2 = (const float*)d_in[5];
  const float* W3 = (const float*)d_in[6];
  const float* b3 = (const float*)d_in[7];
  const float* W4 = (const float*)d_in[8];
  const float* b4 = (const float*)d_in[9];

  float* out     = (float*)d_out;
  float* emb     = out;                                   // [B,G,OUT]
  float* centers = out + (size_t)B_ * G_ * OUT_;          // [B,G,3]
  float* knn_f   = centers + (size_t)B_ * G_ * 3;         // [B,G,K] (as float)

  char* ws = (char*)d_ws;
  int*      fps_i = (int*)ws;                                         // B*G ints
  int*      knn_i = (int*)(ws + 16384);                               // B*G*K ints
  _Float16* wpack = (_Float16*)(ws + 16384 + (size_t)B_ * G_ * K_ * 4);

  // pack weights to f16 fragment layout (716800 elements)
  pack_kernel<<<dim3(2800), dim3(256), 0, stream>>>(W2, W3, W4, wpack);
  // farthest point sampling + centers
  fps_kernel<<<dim3(B_), dim3(1024), 0, stream>>>(coords, fps_i, centers);
  // knn (dynamic LDS: 32768*4 + 256*4 + 256*4 = 133120 B)
  knn_kernel<<<dim3(B_ * G_), dim3(256), 133120, stream>>>(coords, fps_i, knn_i, knn_f);
  // fused WMMA patch encoder (dynamic LDS: 176384 B)
  patch_mlp_kernel<<<dim3(B_ * G_), dim3(256), 176384, stream>>>(
      coords, feats, W1, b1, b2, b3, b4, fps_i, knn_i, wpack, emb);
}
